// GridNetDir_39548058862136
// MI455X (gfx1250) — compile-verified, matrix operand order checked by reading, not run on hardware
//
#include <hip/hip_runtime.h>
#include <math.h>

typedef __attribute__((ext_vector_type(16))) _Float16 v16h;
typedef __attribute__((ext_vector_type(8)))  float    v8f;
typedef __attribute__((ext_vector_type(4)))  _Float16 h4;
typedef __attribute__((ext_vector_type(4)))  float    f4;

#define FEAT          32
#define WAVES_PER_BLK 4
#define PTS_PER_WAVE  16
#define PTS_PER_BLK   (WAVES_PER_BLK * PTS_PER_WAVE)

union AFrag { v16h h; unsigned int u[8]; };
union CFrag { v8f  f; float e[8]; };

__global__ __launch_bounds__(128) void gridnet_fused_kernel(
    const float* __restrict__ pos,      const float* __restrict__ dir,
    const float* __restrict__ grid_pos, const float* __restrict__ grid_dir,
    const float* __restrict__ W1, const float* __restrict__ b1,
    const float* __restrict__ W2, const float* __restrict__ b2,
    float* __restrict__ out, int npts)
{
    // LDS: W1^T (n-major, 128x64 f16) + padded W2^T (16x128 f16)
    //      + per-wave A tile (16x64 f16) + per-wave H tile (16x128 f16)
    __shared__ _Float16 sW1t[128 * 64];                    // 16 KB
    __shared__ _Float16 sW2t[16 * 128];                    //  4 KB
    __shared__ _Float16 sA[WAVES_PER_BLK][16 * 64];        //  8 KB
    __shared__ _Float16 sH[WAVES_PER_BLK][16 * 128];       // 16 KB

    const int tid = threadIdx.x;

    // ---- stage weights (transposed, f32 -> f16) ----
    for (int i = tid; i < 128 * 64; i += 128) {
        int n = i >> 6, k = i & 63;
        sW1t[i] = (_Float16)W1[k * 128 + n];
    }
    for (int i = tid; i < 16 * 128; i += 128) {
        int n = i >> 7, k = i & 127;
        sW2t[i] = (n < 3) ? (_Float16)W2[k * 3 + n] : (_Float16)0.0f;
    }
    __syncthreads();

    const int lane = tid & 31;
    const int wave = tid >> 5;
    const int m    = lane & 15;     // row within tile / N-column for B,C,D
    const int g    = lane >> 4;     // half-wave group
    const long p0  = ((long)blockIdx.x * WAVES_PER_BLK + wave) * PTS_PER_WAVE;

    // ---- bilinear gather: lane computes 32 features of one point ----
    // lanes 0..15 -> grid_pos/pos, lanes 16..31 -> grid_dir/dir
    {
        long pidx = p0 + m;
        if (pidx >= npts) pidx = npts - 1;
        const float* vec  = (g == 0) ? pos : dir;
        const float* grid = (g == 0) ? grid_pos : grid_dir;
        float v0 = vec[2 * pidx + 0];
        float v1 = vec[2 * pidx + 1];
        const float PI = 3.14159265358979323846f;
        float x = v0 * (1023.0f / PI);
        float y = (v1 + PI) * (1023.0f / (2.0f * PI));
        int x0 = (int)x, y0 = (int)y;
        x0 = min(max(x0, 0), 1023);
        y0 = min(max(y0, 0), 1023);
        float xf = x - (float)x0, yf = y - (float)y0;
        int x1 = min(x0 + 1, 1023), y1 = min(y0 + 1, 1023);
        const float* tl = grid + ((long)y0 * 1024 + x0) * FEAT;
        const float* tr = grid + ((long)y0 * 1024 + x1) * FEAT;
        const float* bl = grid + ((long)y1 * 1024 + x0) * FEAT;
        const float* br = grid + ((long)y1 * 1024 + x1) * FEAT;
        float w00 = (1.0f - xf) * (1.0f - yf);
        float w10 = xf * (1.0f - yf);
        float w01 = (1.0f - xf) * yf;
        float w11 = xf * yf;
        _Float16* arow = &sA[wave][m * 64 + g * 32];
        #pragma unroll
        for (int c = 0; c < FEAT; c += 4) {
            f4 a  = *(const f4*)(tl + c);
            f4 b  = *(const f4*)(tr + c);
            f4 cc = *(const f4*)(bl + c);
            f4 d  = *(const f4*)(br + c);
            f4 r  = a * w00 + b * w10 + cc * w01 + d * w11;
            h4 hh;
            hh.x = (_Float16)r.x; hh.y = (_Float16)r.y;
            hh.z = (_Float16)r.z; hh.w = (_Float16)r.w;
            *(h4*)(arow + c) = hh;
        }
    }
    __syncthreads();

    // ---- layer 1: H[16x128] = leaky_relu(A[16x64] @ W1[64x128] + b1) ----
    #pragma unroll
    for (int nt = 0; nt < 8; ++nt) {
        CFrag c;
        float bias = b1[nt * 16 + m];
        #pragma unroll
        for (int r = 0; r < 8; ++r) c.e[r] = bias;
        #pragma unroll
        for (int kc = 0; kc < 2; ++kc) {
            AFrag a, b;
            const _Float16* arow = &sA[wave][m * 64 + kc * 32];
            const _Float16* brow = &sW1t[(nt * 16 + m) * 64 + kc * 32];
            #pragma unroll
            for (int v = 0; v < 8; ++v) {
                int k0 = 2 * (v & 3) + 16 * (v >> 2) + 8 * g;
                a.u[v] = *(const unsigned int*)(arow + k0);
                b.u[v] = *(const unsigned int*)(brow + k0);
            }
            c.f = __builtin_amdgcn_wmma_f32_16x16x32_f16(
                false, a.h, false, b.h, (short)0, c.f, false, false);
        }
        #pragma unroll
        for (int r = 0; r < 8; ++r) {
            int mm = r + 8 * g;
            float h = c.e[r];
            h = (h > 0.0f) ? h : 0.01f * h;
            sH[wave][mm * 128 + nt * 16 + m] = (_Float16)h;
        }
    }
    // wave-local cross-lane RAW through LDS: wait for our DS stores
    asm volatile("s_wait_dscnt 0" ::: "memory");

    // ---- layer 2: out[16x3] = sigmoid(H[16x128] @ W2p[128x16] + b2) * 255 ----
    {
        CFrag c;
        float bias = (m < 3) ? b2[m] : 0.0f;
        #pragma unroll
        for (int r = 0; r < 8; ++r) c.e[r] = bias;
        #pragma unroll
        for (int kc = 0; kc < 4; ++kc) {
            AFrag a, b;
            const _Float16* arow = &sH[wave][m * 128 + kc * 32];
            const _Float16* brow = &sW2t[m * 128 + kc * 32];
            #pragma unroll
            for (int v = 0; v < 8; ++v) {
                int k0 = 2 * (v & 3) + 16 * (v >> 2) + 8 * g;
                a.u[v] = *(const unsigned int*)(arow + k0);
                b.u[v] = *(const unsigned int*)(brow + k0);
            }
            c.f = __builtin_amdgcn_wmma_f32_16x16x32_f16(
                false, a.h, false, b.h, (short)0, c.f, false, false);
        }
        if (m < 3) {
            #pragma unroll
            for (int r = 0; r < 8; ++r) {
                long pidx = p0 + r + 8 * g;
                if (pidx < npts) {
                    float hv = c.e[r];
                    float s  = 1.0f / (1.0f + __expf(-hv));
                    out[pidx * 3 + m] = s * 255.0f;
                }
            }
        }
    }
}

extern "C" void kernel_launch(void* const* d_in, const int* in_sizes, int n_in,
                              void* d_out, int out_size, void* d_ws, size_t ws_size,
                              hipStream_t stream) {
    const float* pos      = (const float*)d_in[0];
    const float* dir      = (const float*)d_in[1];
    const float* grid_pos = (const float*)d_in[2];
    const float* grid_dir = (const float*)d_in[3];
    const float* W1       = (const float*)d_in[4];
    const float* b1       = (const float*)d_in[5];
    const float* W2       = (const float*)d_in[6];
    const float* b2       = (const float*)d_in[7];
    float* out            = (float*)d_out;

    int npts = in_sizes[0] / 2;                 // pos is [N,2]
    int blocks = (npts + PTS_PER_BLK - 1) / PTS_PER_BLK;
    gridnet_fused_kernel<<<blocks, 128, 0, stream>>>(
        pos, dir, grid_pos, grid_dir, W1, b1, W2, b2, out, npts);
}
